// FullHarmonicSystem_26319559590182
// MI455X (gfx1250) — matrix operand order repaired and checked by reading, not compiled, not run on hardware
//
#include <hip/hip_runtime.h>
#include <math.h>

// ---------------------------------------------------------------------------
// FullHarmonicSystem forward for MI455X (gfx1250, wave32, WMMA).
// All large GEMMs run through v_wmma_f32_16x16x32_bf16 (bf16 in, f32 acc).
// ---------------------------------------------------------------------------

typedef __bf16 bf16_t;
typedef __attribute__((ext_vector_type(16))) __bf16 v16bf;
typedef __attribute__((ext_vector_type(8)))  float  v8f;

struct alignas(16) U128 { unsigned int w[4]; };
struct P256 { U128 lo, hi; };   // 32 bytes == sizeof(v16bf)

enum { GF_ACCUM = 1, GF_BIAS = 2, GF_RELU = 4, GF_TANH = 8, GF_ROWSCALE = 16 };

__device__ __forceinline__ bf16_t f2bf(float f) {
  union { float f; unsigned u; } cv; cv.f = f;
  unsigned u = cv.u;
  u += 0x7FFFu + ((u >> 16) & 1u);     // round-to-nearest-even
  union { unsigned short s; bf16_t b; } o; o.s = (unsigned short)(u >> 16);
  return o.b;
}

// ---------------------------------------------------------------------------
// WMMA GEMM: C[M,N](f32) = epilogue( alpha * A[M,K](bf16) * Bt[N,K](bf16)^T )
// Bt is stored N-major (row n holds K contiguous) so both fragments load as
// two ds_load_b128 per lane matching the ISA 16-bit A/B VGPR layouts.
// Tiles: 128x128, BK=32, 8 waves (2x4), wave subtile 64x32 (8 v8f accums).
// Batched via blockIdx.z with zq = z/zmod, zr = z%zmod pointer offsets.
// ---------------------------------------------------------------------------
__global__ __launch_bounds__(256) void gemm_bf16_wmma(
    const bf16_t* __restrict__ A, long lda, long sA1, long sA2,
    const bf16_t* __restrict__ Bt, long ldb, long sB1, long sB2,
    float* __restrict__ C, long ldc, long sC1, long sC2,
    const float* __restrict__ bias,
    const float* __restrict__ rowscale, long rsStride,
    float alpha, int K, int flags, int zmod)
{
  const int tid  = threadIdx.x;
  const int wave = tid >> 5;
  const int lane = tid & 31;
  const long zq = blockIdx.z / zmod;
  const long zr = blockIdx.z % zmod;
  A  += zq * sA1 + zr * sA2;
  Bt += zq * sB1 + zr * sB2;
  C  += zq * sC1 + zr * sC2;

  const int tileM = blockIdx.y * 128;
  const int tileN = blockIdx.x * 128;
  const int wm0 = (wave >> 2) * 64;    // wave row base within tile
  const int wn0 = (wave & 3) * 32;     // wave col base within tile
  const int half16 = lane >> 4;        // 0: lanes 0-15, 1: lanes 16-31
  const int l16 = lane & 15;
  const int kA = half16 ? 8 : 0;       // A frag: K {kA..kA+7} then {kA+16..kA+23}
  const int kB = half16 ? 8 : 0;       // B frag: K {kB..kB+7} then {kB+16..kB+23}

  __shared__ __align__(16) bf16_t As[128][40];   // 80B row stride (16B aligned)
  __shared__ __align__(16) bf16_t Bs[128][40];

  v8f acc[4][2] = {};

  for (int kk = 0; kk < K; kk += 32) {
#pragma unroll
    for (int i = 0; i < 2; ++i) {
      const int chunk = tid + i * 256;       // 512 chunks of 16B = 8KB tile
      const int row = chunk >> 2;
      const int col = (chunk & 3) << 3;
      *reinterpret_cast<U128*>(&As[row][col]) =
          *reinterpret_cast<const U128*>(A + (long)(tileM + row) * lda + (kk + col));
      *reinterpret_cast<U128*>(&Bs[row][col]) =
          *reinterpret_cast<const U128*>(Bt + (long)(tileN + row) * ldb + (kk + col));
    }
    if (kk + 32 < K) {   // pull next K-slab toward L2 (global_prefetch_b8)
      __builtin_prefetch(A + (long)(tileM + (tid >> 1)) * lda + kk + 32, 0, 1);
      __builtin_prefetch(Bt + (long)(tileN + (tid >> 1)) * ldb + kk + 32, 0, 1);
    }
    __syncthreads();

    v16bf af[4], bfr[2];
#pragma unroll
    for (int mi = 0; mi < 4; ++mi) {
      const int r = wm0 + mi * 16 + l16;
      P256 t;
      t.lo = *reinterpret_cast<const U128*>(&As[r][kA]);
      t.hi = *reinterpret_cast<const U128*>(&As[r][kA + 16]);
      af[mi] = __builtin_bit_cast(v16bf, t);
    }
#pragma unroll
    for (int ni = 0; ni < 2; ++ni) {
      const int c = wn0 + ni * 16 + l16;
      P256 t;
      t.lo = *reinterpret_cast<const U128*>(&Bs[c][kB]);
      t.hi = *reinterpret_cast<const U128*>(&Bs[c][kB + 16]);
      bfr[ni] = __builtin_bit_cast(v16bf, t);
    }
#pragma unroll
    for (int mi = 0; mi < 4; ++mi)
#pragma unroll
      for (int ni = 0; ni < 2; ++ni)
        acc[mi][ni] = __builtin_amdgcn_wmma_f32_16x16x32_bf16(
            false, af[mi], false, bfr[ni], (short)0, acc[mi][ni], false, false);
    __syncthreads();
  }

  // Epilogue per C/D layout: lane<16 -> n=l16, rows m..m+7; lane>=16 -> rows m+8..m+15.
#pragma unroll
  for (int mi = 0; mi < 4; ++mi) {
#pragma unroll
    for (int ni = 0; ni < 2; ++ni) {
      const int n = tileN + wn0 + ni * 16 + l16;
      const int mb = tileM + wm0 + mi * 16 + half16 * 8;
      const float bv = (flags & GF_BIAS) ? bias[n] : 0.0f;
#pragma unroll
      for (int r = 0; r < 8; ++r) {
        const int m = mb + r;
        float v = acc[mi][ni][r] * alpha + bv;
        if (flags & GF_TANH) v = tanhf(v);
        if (flags & GF_RELU) v = fmaxf(v, 0.0f);
        if (flags & GF_ROWSCALE) v *= rowscale[(long)m * rsStride];
        float* cp = C + (long)m * ldc + n;
        float o = v;
        if (flags & GF_ACCUM) o += *cp;
        *cp = o;
      }
    }
  }
}

// ---------------------------------------------------------------------------
// Support kernels
// ---------------------------------------------------------------------------
__global__ void convert_bf16_kernel(const float* __restrict__ in,
                                    bf16_t* __restrict__ out, long n) {
  const long stride = (long)gridDim.x * blockDim.x;
  for (long i = (long)blockIdx.x * blockDim.x + threadIdx.x; i < n; i += stride)
    out[i] = f2bf(in[i]);
}

// in[R][C] f32 -> out[C][R] bf16 (LDS-tiled transpose). R, C multiples of 32.
__global__ void convert_transpose_kernel(const float* __restrict__ in,
                                         bf16_t* __restrict__ out, int R, int C) {
  __shared__ float t[32][33];
  const int r0 = blockIdx.y * 32, c0 = blockIdx.x * 32;
  const int tx = threadIdx.x, ty = threadIdx.y;
  for (int i = ty; i < 32; i += 8)
    t[i][tx] = in[(long)(r0 + i) * C + c0 + tx];
  __syncthreads();
  for (int i = ty; i < 32; i += 8)
    out[(long)(c0 + i) * R + r0 + tx] = f2bf(t[tx][i]);
}

// v[B,S,H] f32 -> vt[(b*HEADS+h)][DH][S] bf16  (per-head transpose for probs@V)
__global__ void head_transpose_kernel(const float* __restrict__ in,
                                      bf16_t* __restrict__ out,
                                      int S, int H, int HEADS, int DH) {
  __shared__ float t[32][33];
  const int bh = blockIdx.z;
  const int b = bh / HEADS, h = bh % HEADS;
  const int s0 = blockIdx.x * 32, d0 = blockIdx.y * 32;
  const int tx = threadIdx.x, ty = threadIdx.y;
  for (int i = ty; i < 32; i += 8)
    t[i][tx] = in[((long)b * S + (s0 + i)) * H + h * DH + d0 + tx];
  __syncthreads();
  for (int i = ty; i < 32; i += 8)
    out[((long)bh * DH + (d0 + i)) * S + s0 + tx] = f2bf(t[tx][i]);
}

__global__ __launch_bounds__(256) void layernorm_kernel(
    const float* __restrict__ in, float* __restrict__ out,
    const float* __restrict__ g, const float* __restrict__ b, int C, int relu) {
  const long row = blockIdx.x;
  const float* x = in + row * (long)C;
  float s = 0.f, s2 = 0.f;
  for (int c = threadIdx.x; c < C; c += 256) { const float v = x[c]; s += v; s2 += v * v; }
  __shared__ float rs[256], rq[256];
  rs[threadIdx.x] = s; rq[threadIdx.x] = s2;
  __syncthreads();
  for (int o = 128; o > 0; o >>= 1) {
    if ((int)threadIdx.x < o) { rs[threadIdx.x] += rs[threadIdx.x + o]; rq[threadIdx.x] += rq[threadIdx.x + o]; }
    __syncthreads();
  }
  const float mean = rs[0] / C;
  const float var  = rq[0] / C - mean * mean;
  const float inv  = rsqrtf(var + 1e-5f);
  float* y = out + row * (long)C;
  for (int c = threadIdx.x; c < C; c += 256) {
    float v = (x[c] - mean) * inv * g[c] + b[c];
    if (relu) v = fmaxf(v, 0.f);
    y[c] = v;
  }
}

__global__ void posenc_kernel(float* __restrict__ bx, float* __restrict__ by,
                              long total, int S, int H) {
  const long stride = (long)gridDim.x * blockDim.x;
  for (long i = (long)blockIdx.x * blockDim.x + threadIdx.x; i < total; i += stride) {
    const int j = (int)(i % H);
    const long s = (i / H) % S;
    const float freq = __expf(-(float)j * (9.210340371976184f / (float)H)); // 10000^(-j/H)
    const float ang = (float)s * freq;
    bx[i] += __cosf(ang);
    by[i] += __sinf(ang);
  }
}

// g = relu(cos(atan2(ny,nx))) = relu(nx / |n|);  b{x,y} += 0.1 * n{x,y} * g
__global__ void phasegate_kernel(const float* __restrict__ nx, const float* __restrict__ ny,
                                 float* __restrict__ bx, float* __restrict__ by, long total) {
  const long stride = (long)gridDim.x * blockDim.x;
  for (long i = (long)blockIdx.x * blockDim.x + threadIdx.x; i < total; i += stride) {
    const float a = nx[i], b = ny[i];
    const float g = fmaxf(a * rsqrtf(a * a + b * b + 1e-30f), 0.f);
    bx[i] += 0.1f * a * g;
    by[i] += 0.1f * b * g;
  }
}

__global__ void mag_kernel(const float* __restrict__ hx, const float* __restrict__ hy,
                           float* __restrict__ mag, long total) {
  const long stride = (long)gridDim.x * blockDim.x;
  for (long i = (long)blockIdx.x * blockDim.x + threadIdx.x; i < total; i += stride) {
    const float a = hx[i], b = hy[i];
    mag[i] = sqrtf(a * a + b * b + 1e-8f);
  }
}

// Row softmax (len C), output bf16 for the following WMMA GEMM.
__global__ __launch_bounds__(256) void softmax_kernel(const float* __restrict__ in,
                                                      bf16_t* __restrict__ out, int C) {
  const long row = blockIdx.x;
  const float* x = in + row * (long)C;
  __shared__ float red[256];
  float mx = -3.4e38f;
  for (int c = threadIdx.x; c < C; c += 256) mx = fmaxf(mx, x[c]);
  red[threadIdx.x] = mx; __syncthreads();
  for (int o = 128; o > 0; o >>= 1) {
    if ((int)threadIdx.x < o) red[threadIdx.x] = fmaxf(red[threadIdx.x], red[threadIdx.x + o]);
    __syncthreads();
  }
  mx = red[0];
  __syncthreads();
  float s = 0.f;
  for (int c = threadIdx.x; c < C; c += 256) s += __expf(x[c] - mx);
  red[threadIdx.x] = s; __syncthreads();
  for (int o = 128; o > 0; o >>= 1) {
    if ((int)threadIdx.x < o) red[threadIdx.x] += red[threadIdx.x + o];
    __syncthreads();
  }
  const float inv = 1.f / red[0];
  bf16_t* y = out + row * (long)C;
  for (int c = threadIdx.x; c < C; c += 256) y[c] = f2bf(__expf(x[c] - mx) * inv);
}

// gate[row, 0..7] = softmax_e( mag[row,:] . Wg[:,e] + bg[e] ), one wave per row.
__global__ __launch_bounds__(256) void gate_kernel(const float* __restrict__ mag,
                                                   const float* __restrict__ Wg,
                                                   const float* __restrict__ bg,
                                                   float* __restrict__ gate,
                                                   int Kd, int E) {
  const int wave = threadIdx.x >> 5, lane = threadIdx.x & 31;
  const long row = (long)blockIdx.x * 8 + wave;
  const float* m = mag + row * Kd;
  float acc[8] = {0.f, 0.f, 0.f, 0.f, 0.f, 0.f, 0.f, 0.f};
  for (int k = lane; k < Kd; k += 32) {
    const float mv = m[k];
#pragma unroll
    for (int e = 0; e < 8; ++e) acc[e] += mv * Wg[(long)k * E + e];
  }
#pragma unroll
  for (int off = 16; off > 0; off >>= 1)
#pragma unroll
    for (int e = 0; e < 8; ++e) acc[e] += __shfl_xor(acc[e], off);
  if (lane == 0) {
    float v[8], mx = -3.4e38f;
#pragma unroll
    for (int e = 0; e < 8; ++e) { v[e] = acc[e] + bg[e]; mx = fmaxf(mx, v[e]); }
    float s = 0.f;
#pragma unroll
    for (int e = 0; e < 8; ++e) { v[e] = __expf(v[e] - mx); s += v[e]; }
    const float inv = 1.f / s;
#pragma unroll
    for (int e = 0; e < 8; ++e) gate[row * E + e] = v[e] * inv;
  }
}

// ---------------------------------------------------------------------------
// Host orchestration
// ---------------------------------------------------------------------------
extern "C" void kernel_launch(void* const* d_in, const int* in_sizes, int n_in,
                              void* d_out, int out_size, void* d_ws, size_t ws_size,
                              hipStream_t stream) {
  (void)in_sizes; (void)n_in; (void)out_size; (void)ws_size;
  constexpr int B = 16, S = 512, IN = 512, H = 1024, OUT = 512;
  constexpr int HEADS = 8, HUB = 1024, SPEC = 8, DEPTH = 3;
  constexpr int DH = H / HEADS;            // 128
  constexpr long NT = (long)B * S;         // 8192
  constexpr int BH = B * HEADS;            // 128

  auto F = [&](int i) { return (const float*)d_in[i]; };
  const float* x          = F(0);
  const float* enc_prep_w = F(1);  const float* enc_prep_b = F(2);
  const float* enc_ln_g   = F(3);  const float* enc_ln_b   = F(4);
  const float* enc_x_w    = F(5);  const float* enc_x_b    = F(6);
  const float* enc_y_w    = F(7);  const float* enc_y_b    = F(8);
  const float* ph_wr[DEPTH]; const float* ph_wi[DEPTH];
  const float* ph_bx[DEPTH]; const float* ph_by[DEPTH];
  const float* ph_gx[DEPTH]; const float* ph_ox[DEPTH];
  const float* ph_gy[DEPTH]; const float* ph_oy[DEPTH];
  for (int l = 0; l < DEPTH; ++l) {
    const int bidx = 9 + 8 * l;
    ph_wr[l] = F(bidx + 0); ph_wi[l] = F(bidx + 1);
    ph_bx[l] = F(bidx + 2); ph_by[l] = F(bidx + 3);
    ph_gx[l] = F(bidx + 4); ph_ox[l] = F(bidx + 5);
    ph_gy[l] = F(bidx + 6); ph_oy[l] = F(bidx + 7);
  }
  const float* att_q_w = F(33); const float* att_q_b = F(34);
  const float* att_k_w = F(35); const float* att_k_b = F(36);
  const float* att_v_w = F(37); const float* att_v_b = F(38);
  const float* att_o_w = F(39); const float* att_o_b = F(40);
  const float* hubx_w  = F(41); const float* hubx_b  = F(42);
  const float* huby_w  = F(43); const float* huby_b  = F(44);
  const float* gate_w  = F(45); const float* gate_b  = F(46);
  const float* spec_w  = F(47); const float* spec_b  = F(48);
  const float* dec_x_w = F(49); const float* dec_x_b = F(50);
  const float* dec_y_w = F(51); const float* dec_y_b = F(52);
  const float* dec_m_w = F(53); const float* dec_m_b = F(54);

  // Bump allocator over d_ws (deterministic; same pointers every call).
  char* base = (char*)d_ws;
  size_t off = 0;
  auto alloc  = [&](size_t bytes) -> void* {
    off = (off + 255) & ~(size_t)255;
    void* p = base + off; off += bytes; return p;
  };
  auto allocF = [&](long n) { return (float*)alloc((size_t)n * 4); };
  auto allocB = [&](long n) { return (bf16_t*)alloc((size_t)n * 2); };

  // bf16 weights (Bt layout: [N][K])
  bf16_t* wPrepT = allocB((long)H * IN);
  bf16_t* wEncXT = allocB((long)H * H);
  bf16_t* wEncYT = allocB((long)H * H);
  bf16_t* wPhRT[DEPTH]; bf16_t* wPhIT[DEPTH];
  for (int l = 0; l < DEPTH; ++l) { wPhRT[l] = allocB((long)H * H); wPhIT[l] = allocB((long)H * H); }
  bf16_t* wQT = allocB((long)H * H); bf16_t* wKT = allocB((long)H * H);
  bf16_t* wVT = allocB((long)H * H); bf16_t* wOT = allocB((long)H * H);
  bf16_t* wHubXT = allocB((long)HUB * H);  bf16_t* wHubYT = allocB((long)HUB * H);
  bf16_t* wHubX  = allocB((long)H * HUB);  bf16_t* wHubY  = allocB((long)H * HUB);
  bf16_t* wSpecT = allocB((long)SPEC * HUB * HUB);
  bf16_t* wDecXT = allocB((long)OUT * H);  bf16_t* wDecYT = allocB((long)OUT * H);
  bf16_t* wDecMT = allocB((long)OUT * 2 * OUT);

  // Activations (with aliasing noted)
  bf16_t* xbf  = allocB(NT * IN);
  float*  t0   = allocF(NT * H);          // pre-LN -> h (LN in place)
  bf16_t* hbf  = allocB(NT * H);          // h bf16; later reused as mag_bf
  float*  bx   = allocF(NT * H);          // later: attention context x
  float*  by   = allocF(NT * H);          // later: attention context y
  float*  tx   = allocF(NT * H);          // phase tmp; later ox / fx
  float*  ty   = allocF(NT * H);          // phase tmp; later oy / fy
  bf16_t* abf0 = allocB(NT * H);          // rotating bf16 staging
  bf16_t* abf1 = allocB(NT * H);
  float*  qx   = allocF(NT * H);          // later hx
  float*  qy   = allocF(NT * H);          // later hy
  float*  kx   = allocF(NT * H);          // later mag
  float*  ky   = allocF(NT * H);          // later hub
  float*  vx   = allocF(NT * H);          // later dx
  float*  vy   = allocF(NT * H);          // later dy
  bf16_t* vtx  = allocB((long)BH * DH * S);
  bf16_t* vty  = allocB((long)BH * DH * S);
  float*  scores = allocF((long)BH * S * S);
  bf16_t* probs  = allocB((long)BH * S * S);
  float*  gate   = allocF(NT * SPEC);
  float*  out    = (float*)d_out;

  // Helpers -----------------------------------------------------------------
  auto conv = [&](const float* src, bf16_t* dst, long n) {
    unsigned blocks = (unsigned)((n + 255) / 256);
    if (blocks > 4096u) blocks = 4096u;
    convert_bf16_kernel<<<dim3(blocks), 256, 0, stream>>>(src, dst, n);
  };
  auto convT = [&](const float* src, bf16_t* dst, int R, int C) {
    convert_transpose_kernel<<<dim3(C / 32, R / 32), dim3(32, 8), 0, stream>>>(src, dst, R, C);
  };
  auto gemm = [&](const bf16_t* A, long lda, long sA1, long sA2,
                  const bf16_t* Bt, long ldb, long sB1, long sB2,
                  float* C, long ldc, long sC1, long sC2,
                  const float* bias, const float* rowscale, long rsStride,
                  float alpha, int M, int N, int K, int flags, int batches, int zmod) {
    dim3 g((unsigned)(N / 128), (unsigned)(M / 128), (unsigned)batches);
    gemm_bf16_wmma<<<g, 256, 0, stream>>>(A, lda, sA1, sA2, Bt, ldb, sB1, sB2,
                                          C, ldc, sC1, sC2, bias, rowscale, rsStride,
                                          alpha, K, flags, zmod);
  };
  auto gemm1 = [&](const bf16_t* A, long lda, const bf16_t* Bt, long ldb,
                   float* C, long ldc, const float* bias,
                   float alpha, int M, int N, int K, int flags) {
    gemm(A, lda, 0, 0, Bt, ldb, 0, 0, C, ldc, 0, 0, bias, nullptr, 0,
         alpha, M, N, K, flags, 1, 1);
  };
  auto ln = [&](const float* in, float* outp, const float* g, const float* b, int relu) {
    layernorm_kernel<<<dim3((unsigned)NT), 256, 0, stream>>>(in, outp, g, b, H, relu);
  };

  // Weight conversion -------------------------------------------------------
  convT(enc_prep_w, wPrepT, IN, H);
  convT(enc_x_w, wEncXT, H, H);
  convT(enc_y_w, wEncYT, H, H);
  for (int l = 0; l < DEPTH; ++l) { convT(ph_wr[l], wPhRT[l], H, H); convT(ph_wi[l], wPhIT[l], H, H); }
  convT(att_q_w, wQT, H, H); convT(att_k_w, wKT, H, H);
  convT(att_v_w, wVT, H, H); convT(att_o_w, wOT, H, H);
  convT(hubx_w, wHubXT, H, HUB); convT(huby_w, wHubYT, H, HUB);
  conv(hubx_w, wHubX, (long)H * HUB);   // plain: Bt for back-projection (hub @ W^T)
  conv(huby_w, wHubY, (long)H * HUB);
  for (int e = 0; e < SPEC; ++e)
    convT(spec_w + (long)e * HUB * HUB, wSpecT + (long)e * HUB * HUB, HUB, HUB);
  convT(dec_x_w, wDecXT, H, OUT); convT(dec_y_w, wDecYT, H, OUT);
  convT(dec_m_w, wDecMT, 2 * OUT, OUT);

  // Encoder -----------------------------------------------------------------
  conv(x, xbf, NT * IN);
  gemm1(xbf, IN, wPrepT, IN, t0, H, enc_prep_b, 1.f, (int)NT, H, IN, GF_BIAS);
  ln(t0, t0, enc_ln_g, enc_ln_b, 1);            // LN + ReLU -> h (in place)
  conv(t0, hbf, NT * H);
  gemm1(hbf, H, wEncXT, H, bx, H, enc_x_b, 1.f, (int)NT, H, H, GF_BIAS);
  gemm1(hbf, H, wEncYT, H, by, H, enc_y_b, 1.f, (int)NT, H, H, GF_BIAS);
  posenc_kernel<<<dim3(4096), 256, 0, stream>>>(bx, by, NT * H, S, H);

  // Phase layers ------------------------------------------------------------
  for (int l = 0; l < DEPTH; ++l) {
    conv(bx, abf0, NT * H);
    conv(by, abf1, NT * H);
    gemm1(abf0, H, wPhRT[l], H, tx, H, ph_bx[l],  1.f, (int)NT, H, H, GF_BIAS);
    gemm1(abf1, H, wPhIT[l], H, tx, H, nullptr,  -1.f, (int)NT, H, H, GF_ACCUM);
    gemm1(abf0, H, wPhIT[l], H, ty, H, ph_by[l],  1.f, (int)NT, H, H, GF_BIAS);
    gemm1(abf1, H, wPhRT[l], H, ty, H, nullptr,   1.f, (int)NT, H, H, GF_ACCUM);
    ln(tx, tx, ph_gx[l], ph_ox[l], 0);
    ln(ty, ty, ph_gy[l], ph_oy[l], 0);
    phasegate_kernel<<<dim3(4096), 256, 0, stream>>>(tx, ty, bx, by, NT * H);
  }

  // Attention ---------------------------------------------------------------
  conv(bx, abf0, NT * H);
  conv(by, abf1, NT * H);
  gemm1(abf0, H, wQT, H, qx, H, att_q_b, 1.f, (int)NT, H, H, GF_BIAS);
  gemm1(abf1, H, wQT, H, qy, H, att_q_b, 1.f, (int)NT, H, H, GF_BIAS);
  gemm1(abf0, H, wKT, H, kx, H, att_k_b, 1.f, (int)NT, H, H, GF_BIAS);
  gemm1(abf1, H, wKT, H, ky, H, att_k_b, 1.f, (int)NT, H, H, GF_BIAS);
  gemm1(abf0, H, wVT, H, vx, H, att_v_b, 1.f, (int)NT, H, H, GF_BIAS);
  gemm1(abf1, H, wVT, H, vy, H, att_v_b, 1.f, (int)NT, H, H, GF_BIAS);
  head_transpose_kernel<<<dim3(S / 32, DH / 32, BH), dim3(32, 8), 0, stream>>>(vx, vtx, S, H, HEADS, DH);
  head_transpose_kernel<<<dim3(S / 32, DH / 32, BH), dim3(32, 8), 0, stream>>>(vy, vty, S, H, HEADS, DH);

  const float sc = 0.08838834764831843f;  // 1/sqrt(DH)
  // scores = (qx.kx + qy.ky) / sqrt(DH), batched over (b,h)
  conv(qx, abf0, NT * H);
  conv(kx, abf1, NT * H);
  gemm(abf0, H, (long)S * H, DH, abf1, H, (long)S * H, DH,
       scores, S, (long)HEADS * S * S, (long)S * S,
       nullptr, nullptr, 0, sc, S, S, DH, 0, BH, HEADS);
  conv(qy, abf0, NT * H);
  conv(ky, abf1, NT * H);
  gemm(abf0, H, (long)S * H, DH, abf1, H, (long)S * H, DH,
       scores, S, (long)HEADS * S * S, (long)S * S,
       nullptr, nullptr, 0, sc, S, S, DH, GF_ACCUM, BH, HEADS);
  softmax_kernel<<<dim3((unsigned)((long)BH * S)), 256, 0, stream>>>(scores, probs, S);

  // context: ctx_x -> bx, ctx_y -> by   (probs @ V, batched)
  gemm(probs, S, (long)HEADS * S * S, (long)S * S, vtx, S, (long)HEADS * DH * S, (long)DH * S,
       bx, H, (long)S * H, DH, nullptr, nullptr, 0, 1.f, S, DH, S, 0, BH, HEADS);
  gemm(probs, S, (long)HEADS * S * S, (long)S * S, vty, S, (long)HEADS * DH * S, (long)DH * S,
       by, H, (long)S * H, DH, nullptr, nullptr, 0, 1.f, S, DH, S, 0, BH, HEADS);

  // output projection: ox -> tx, oy -> ty
  conv(bx, abf0, NT * H);
  gemm1(abf0, H, wOT, H, tx, H, att_o_b, 1.f, (int)NT, H, H, GF_BIAS);
  conv(by, abf1, NT * H);
  gemm1(abf1, H, wOT, H, ty, H, att_o_b, 1.f, (int)NT, H, H, GF_BIAS);

  // Hub ---------------------------------------------------------------------
  conv(tx, abf0, NT * H);
  gemm1(abf0, H, wHubXT, H, qx, HUB, hubx_b, 1.f, (int)NT, HUB, H, GF_BIAS);  // hx
  conv(ty, abf1, NT * H);
  gemm1(abf1, H, wHubYT, H, qy, HUB, huby_b, 1.f, (int)NT, HUB, H, GF_BIAS);  // hy
  mag_kernel<<<dim3(4096), 256, 0, stream>>>(qx, qy, kx, NT * HUB);           // mag -> kx
  conv(kx, hbf, NT * HUB);                                                    // mag_bf
  gate_kernel<<<dim3((unsigned)(NT / 8)), 256, 0, stream>>>(kx, gate_w, gate_b, gate, HUB, SPEC);
  for (int e = 0; e < SPEC; ++e) {   // hub(ky) (+)= gate[:,e] * tanh(mag @ We + be)
    gemm(hbf, HUB, 0, 0, wSpecT + (long)e * HUB * HUB, HUB, 0, 0,
         ky, HUB, 0, 0, spec_b + (long)e * HUB, gate + e, SPEC,
         1.f, (int)NT, HUB, HUB,
         GF_BIAS | GF_TANH | GF_ROWSCALE | (e ? GF_ACCUM : 0), 1, 1);
  }
  conv(ky, abf0, NT * HUB);                                                   // hub_bf
  gemm1(abf0, HUB, wHubX, HUB, tx, H, nullptr, 0.3f, (int)NT, H, HUB, GF_ACCUM);  // fx
  gemm1(abf0, HUB, wHubY, HUB, ty, H, nullptr, 0.3f, (int)NT, H, HUB, GF_ACCUM);  // fy

  // Decoder -----------------------------------------------------------------
  conv(tx, abf1, NT * H);
  gemm1(abf1, H, wDecXT, H, vx, OUT, dec_x_b, 1.f, (int)NT, OUT, H, GF_BIAS); // dx
  conv(ty, abf1, NT * H);
  gemm1(abf1, H, wDecYT, H, vy, OUT, dec_y_b, 1.f, (int)NT, OUT, H, GF_BIAS); // dy
  conv(vx, abf0, NT * OUT);
  gemm1(abf0, OUT, wDecMT, 2 * OUT, out, OUT, dec_m_b, 1.f, (int)NT, OUT, OUT, GF_BIAS);
  conv(vy, abf1, NT * OUT);
  gemm1(abf1, OUT, wDecMT + OUT, 2 * OUT, out, OUT, nullptr, 1.f, (int)NT, OUT, OUT, GF_ACCUM);
}
